// SSA_87840671137906
// MI455X (gfx1250) — matrix operand order repaired and checked by reading, not compiled, not run on hardware
//
#include <hip/hip_runtime.h>
#include <hip/hip_bf16.h>
#include <stdint.h>

typedef __attribute__((ext_vector_type(16))) _Float16 v16h;
typedef __attribute__((ext_vector_type(8)))  float    v8f;
typedef __attribute__((ext_vector_type(4)))  unsigned u32x4;
typedef __attribute__((ext_vector_type(8)))  int      i32x8;
typedef __attribute__((ext_vector_type(4)))  int      i32x4;

#define B_   2
#define T_   2048
#define C_   768
#define H_   12
#define HS_  64
#define SCALE_ 0.036084391824351615f   /* 1/sqrt(768) */

// ---------------------------------------------------------------------------
// Fragment loaders (wave32, 16-bit A 16x32 / B 32x16), contiguous k = 16*(lane/16) + s
// ---------------------------------------------------------------------------

// A-matrix 16x32 (MxK), element (m,k) at base[m*ld + k].
__device__ inline v16h frag_a_load(const _Float16* base, int ld) {
  const int lane = threadIdx.x & 31;
  const int row  = lane & 15;
  const int g    = lane >> 4;
  const _Float16* p = base + row * ld;
  union { v16h v; unsigned u[8]; } f;
#pragma unroll
  for (int i = 0; i < 8; ++i)
    f.u[i] = *(const unsigned*)(p + g * 16 + i * 2);
  return f.v;
}

// B-matrix 32x16 (KxN), element (k,n) at base[n*ld + k] (torch W rows are columns)
__device__ inline v16h frag_b_load(const _Float16* base, int ld) {
  const int lane = threadIdx.x & 31;
  const int n    = lane & 15;
  const int g    = lane >> 4;
  const _Float16* p = base + n * ld + g * 16;
  union { v16h v; unsigned u[8]; } f;
#pragma unroll
  for (int i = 0; i < 8; ++i)
    f.u[i] = *(const unsigned*)(p + i * 2);
  return f.v;
}

// B-matrix 32x16 (KxN), element (k,n) at base[k*ld + n] (V tile in LDS)
__device__ inline v16h frag_b_colmajor(const _Float16* base, int ld) {
  const int lane = threadIdx.x & 31;
  const int n    = lane & 15;
  const int g    = lane >> 4;
  union { v16h v; unsigned u[8]; } f;
#pragma unroll
  for (int i = 0; i < 8; ++i) {
    const int k0 = g * 16 + i * 2;
    union { unsigned u; _Float16 h[2]; } t;
    t.h[0] = base[(size_t)k0 * ld + n];
    t.h[1] = base[(size_t)(k0 + 1) * ld + n];
    f.u[i] = t.u;
  }
  return f.v;
}

// ---------------------------------------------------------------------------
// TDM: async 2D f16 tile (tile_h rows x tile_w cols, row stride in elements)
// from global into LDS at byte offset lds_byte_off. D# per ISA 08 §8.
// 6-arg builtin form (clang-23 / therock-10.0 headers).
// ---------------------------------------------------------------------------
__device__ inline void tdm_load_2d_f16(unsigned lds_byte_off, const _Float16* gptr,
                                       int tile_w, int tile_h, int row_stride) {
  const unsigned long long ga = (unsigned long long)(uintptr_t)gptr;
  u32x4 g0;
  g0[0] = 1u;                                                 // count=1 (valid user D#)
  g0[1] = lds_byte_off;                                       // lds_addr [63:32]
  g0[2] = (unsigned)(ga & 0xFFFFFFFFu);                       // global_addr [95:64]
  g0[3] = (unsigned)((ga >> 32) & 0x01FFFFFFu) | (2u << 30);  // addr[56:32] | type=2
  i32x8 g1;
  g1[0] = (1 << 16);                                          // data_size=1 (2 bytes)
  g1[1] = (int)((tile_w & 0xFFFF) << 16);                     // tensor_dim0 [79:48] lo
  g1[2] = (int)(((unsigned)tile_w >> 16) |
                ((tile_h & 0xFFFF) << 16));                   // td0 hi | tensor_dim1 lo
  g1[3] = (int)(((unsigned)tile_h >> 16) |
                ((tile_w & 0xFFFF) << 16));                   // td1 hi | tile_dim0
  g1[4] = (int)(tile_h & 0xFFFF);                             // tile_dim1 (tile_dim2=0)
  g1[5] = row_stride;                                         // tensor_dim0_stride lo
  g1[6] = 0;                                                  // stride hi | dim1_stride lo
  g1[7] = 0;
  i32x4 gz4 = {0, 0, 0, 0};
  i32x8 gz8 = {0, 0, 0, 0, 0, 0, 0, 0};
  __builtin_amdgcn_tensor_load_to_lds(g0, g1, gz4, gz4, gz8, 0);
}

// ---------------------------------------------------------------------------
// fp32 -> fp16 convert
// ---------------------------------------------------------------------------
__global__ void cvt_f32_f16_kernel(const float* __restrict__ in,
                                   _Float16* __restrict__ out, int n) {
  int i = blockIdx.x * blockDim.x + threadIdx.x;
  if (i < n) out[i] = (_Float16)in[i];
}

// ---------------------------------------------------------------------------
// GEMM: out[m,n] = sum_k A[m,k] * W[n,k] + bias[n].  K must be a multiple of 64.
// Block = 128 threads (4 waves); each wave does a 32x64 tile (8 WMMA/k-step).
// k-loop unrolled x2 with ping-pong register sets: no shadow copies, loads of
// one set overlap WMMAs of the other. Grid (M/128, N/64).
// ---------------------------------------------------------------------------
__global__ __launch_bounds__(128)
void wmma_gemm_kernel(const _Float16* __restrict__ A,
                      const _Float16* __restrict__ W,
                      const float* __restrict__ bias,
                      _Float16* __restrict__ outH,
                      float* __restrict__ outF,
                      int M, int N, int K) {
  const int w    = threadIdx.x >> 5;
  const int lane = threadIdx.x & 31;
  const int m0   = blockIdx.x * 128 + w * 32;
  const int n0   = blockIdx.y * 64;

  const _Float16* Abase = A + (size_t)m0 * K;

  v8f acc[2][4] = {};
  v16h a0[2], b0[4], a1[2], b1[4];

#pragma unroll
  for (int mi = 0; mi < 2; ++mi) a0[mi] = frag_a_load(Abase + (size_t)(mi * 16) * K, K);
#pragma unroll
  for (int t = 0; t < 4; ++t)   b0[t] = frag_b_load(W + (size_t)(n0 + t * 16) * K, K);

#pragma unroll 1
  for (int k = 0; k < K; k += 64) {
    // load k+32 into set1 (always in range: K % 64 == 0)
#pragma unroll
    for (int mi = 0; mi < 2; ++mi)
      a1[mi] = frag_a_load(Abase + (size_t)(mi * 16) * K + k + 32, K);
#pragma unroll
    for (int t = 0; t < 4; ++t)
      b1[t] = frag_b_load(W + (size_t)(n0 + t * 16) * K + k + 32, K);
    if (k + 128 < K) {  // pull k+128 toward L2 (global_prefetch_b8)
      __builtin_prefetch(Abase + (size_t)(lane & 15) * K + k + 128, 0, 1);
      __builtin_prefetch(W + (size_t)(n0 + (lane & 15)) * K + k + 128, 0, 1);
    }
    // compute on set0 while set1 loads are in flight
#pragma unroll
    for (int mi = 0; mi < 2; ++mi)
#pragma unroll
      for (int t = 0; t < 4; ++t)
        acc[mi][t] = __builtin_amdgcn_wmma_f32_16x16x32_f16(
            false, a0[mi], false, b0[t], (short)0, acc[mi][t], false, false);
    // load k+64 into set0 while computing on set1
    if (k + 64 < K) {
#pragma unroll
      for (int mi = 0; mi < 2; ++mi)
        a0[mi] = frag_a_load(Abase + (size_t)(mi * 16) * K + k + 64, K);
#pragma unroll
      for (int t = 0; t < 4; ++t)
        b0[t] = frag_b_load(W + (size_t)(n0 + t * 16) * K + k + 64, K);
    }
#pragma unroll
    for (int mi = 0; mi < 2; ++mi)
#pragma unroll
      for (int t = 0; t < 4; ++t)
        acc[mi][t] = __builtin_amdgcn_wmma_f32_16x16x32_f16(
            false, a1[mi], false, b1[t], (short)0, acc[mi][t], false, false);
  }

  const int half = lane >> 4;
  const int nn   = lane & 15;
#pragma unroll
  for (int mi = 0; mi < 2; ++mi)
#pragma unroll
    for (int t = 0; t < 4; ++t)
#pragma unroll
      for (int r = 0; r < 8; ++r) {
        const int m = m0 + mi * 16 + r + 8 * half;
        const int n = n0 + t * 16 + nn;
        const float v = acc[mi][t][r] + bias[n];
        if (outH) outH[(size_t)m * N + n] = (_Float16)v;
        if (outF) outF[(size_t)m * N + n] = v;
      }
}

// ---------------------------------------------------------------------------
// Flash attention (causal, online softmax). Block = 128 threads = 4 waves;
// block handles one (b, h, 64-query tile); each wave owns 16 query rows.
// K/V tiles staged by the Tensor Data Mover into double-buffered LDS:
// wave 0 issues tile kt+1 into the alternate buffer, then waits
// tensorcnt<=2 (TDM ops of one wave complete in order => tile kt is ready).
// LDS: 2x(4KB K + 4KB V) ping-pong + 4x1KB P staging = 20KB.
// ---------------------------------------------------------------------------
__global__ __launch_bounds__(128)
void attn_kernel(const _Float16* __restrict__ xh,   // keys  [B,T,C]
                 const _Float16* __restrict__ uh,   // query [B,T,C]
                 const _Float16* __restrict__ vh,   // value [B,T,C]
                 _Float16* __restrict__ yh) {       // out   [B,T,C]
  // halves: [K0 2048][V0 2048][K1 2048][V1 2048][P 4*512]
  __shared__ __align__(16) _Float16 smem[8192 + 2048];

  const int ntq = T_ / 64;
  const int bid = blockIdx.x;
  const int qt  = bid % ntq;
  const int h   = (bid / ntq) % H_;
  const int b   = bid / (ntq * H_);

  const int w     = threadIdx.x >> 5;
  const int lane  = threadIdx.x & 31;
  const int half  = lane >> 4;
  const int nn    = lane & 15;
  const int qbase = qt * 64;
  const int q0    = qbase + w * 16;

  _Float16* lds_pw = smem + 8192 + w * 512;

  const float NEG_INF = -__builtin_inff();

  // Q fragments (16 rows x 64 hs), loaded once from global.
  const _Float16* qptr = uh + ((size_t)(b * T_ + q0)) * C_ + h * HS_;
  const v16h qf0 = frag_a_load(qptr, C_);
  const v16h qf1 = frag_a_load(qptr + 32, C_);

  v8f o[4] = {};
  float mrow[8], lrow[8];
#pragma unroll
  for (int r = 0; r < 8; ++r) { mrow[r] = NEG_INF; lrow[r] = 0.f; }

  const int ktiles = qbase / 32 + 2;  // cover keys 0 .. qbase+63

  // prologue: wave 0 launches TDM for tile 0 into buffer 0
  if (w == 0) {
    const size_t goff0 = ((size_t)(b * T_)) * C_ + h * HS_;
    tdm_load_2d_f16(0u,    xh + goff0, HS_, 32, C_);
    tdm_load_2d_f16(4096u, vh + goff0, HS_, 32, C_);
  }

#pragma unroll 1
  for (int kt = 0; kt < ktiles; ++kt) {
    const int cur   = kt & 1;
    const int kbase = kt * 32;
    __syncthreads();  // prior iteration's readers of buf[cur^1] are done
    if (w == 0) {
      const bool more = (kt + 1) < ktiles;
      if (more) {  // launch tile kt+1 into the alternate buffer
        const size_t goff = ((size_t)(b * T_ + kbase + 32)) * C_ + h * HS_;
        const unsigned kb = (cur ^ 1) ? 8192u : 0u;
        tdm_load_2d_f16(kb,         xh + goff, HS_, 32, C_);
        tdm_load_2d_f16(kb + 4096u, vh + goff, HS_, 32, C_);
        __builtin_amdgcn_s_wait_tensorcnt(2);  // drain tile kt's pair
      } else {
        __builtin_amdgcn_s_wait_tensorcnt(0);
      }
    }
    __syncthreads();  // tile kt visible to all waves

    const _Float16* lds_k = smem + cur * 4096;
    const _Float16* lds_v = lds_k + 2048;

    // S = Q * K^T : two 16x16 tiles (keys kbase..+15, +16..+31), hs = 2 k-steps
    v8f s[2] = {};
#pragma unroll
    for (int f = 0; f < 2; ++f) {
      v16h bk0 = frag_b_load(lds_k + (f * 16) * 64, 64);
      s[f] = __builtin_amdgcn_wmma_f32_16x16x32_f16(
          false, qf0, false, bk0, (short)0, s[f], false, false);
      v16h bk1 = frag_b_load(lds_k + (f * 16) * 64 + 32, 64);
      s[f] = __builtin_amdgcn_wmma_f32_16x16x32_f16(
          false, qf1, false, bk1, (short)0, s[f], false, false);
    }

    // scale, causal mask, online softmax (rows live per 16-lane half)
#pragma unroll
    for (int r = 0; r < 8; ++r) {
      const int q = q0 + r + 8 * half;
      float s0 = s[0][r] * SCALE_;
      float s1 = s[1][r] * SCALE_;
      if (kbase + nn > q)      s0 = NEG_INF;
      if (kbase + 16 + nn > q) s1 = NEG_INF;

      float rmax = fmaxf(s0, s1);
#pragma unroll
      for (int off = 1; off < 16; off <<= 1)
        rmax = fmaxf(rmax, __shfl_xor(rmax, off, 16));

      const float newm = fmaxf(mrow[r], rmax);
      float p0, p1, alpha;
      if (newm > NEG_INF) {
        alpha = __expf(mrow[r] - newm);
        p0 = (s0 > NEG_INF) ? __expf(s0 - newm) : 0.f;
        p1 = (s1 > NEG_INF) ? __expf(s1 - newm) : 0.f;
      } else {
        alpha = 1.f; p0 = 0.f; p1 = 0.f;
      }
      float rsum = p0 + p1;
#pragma unroll
      for (int off = 1; off < 16; off <<= 1)
        rsum += __shfl_xor(rsum, off, 16);

      mrow[r] = newm;
      lrow[r] = lrow[r] * alpha + rsum;
#pragma unroll
      for (int t = 0; t < 4; ++t) o[t][r] = o[t][r] * alpha;

      const int m = r + 8 * half;
      lds_pw[m * 32 + nn]      = (_Float16)p0;
      lds_pw[m * 32 + 16 + nn] = (_Float16)p1;
    }
    __syncthreads();  // uniform across block (ktiles identical)

    // O += P (16x32) * V (32x64)
    v16h pf = frag_a_load(lds_pw, 32);
#pragma unroll
    for (int t = 0; t < 4; ++t) {
      v16h vf = frag_b_colmajor(lds_v + t * 16, 64);
      o[t] = __builtin_amdgcn_wmma_f32_16x16x32_f16(
          false, pf, false, vf, (short)0, o[t], false, false);
    }
  }

  // epilogue: normalize and write y (f16 for the projection GEMM)
#pragma unroll
  for (int r = 0; r < 8; ++r) {
    const int q     = q0 + r + 8 * half;
    const float inv = 1.f / lrow[r];
#pragma unroll
    for (int t = 0; t < 4; ++t) {
      yh[((size_t)(b * T_ + q)) * C_ + h * HS_ + t * 16 + nn] =
          (_Float16)(o[t][r] * inv);
    }
  }
}

// ---------------------------------------------------------------------------
// Host launcher
// ---------------------------------------------------------------------------
extern "C" void kernel_launch(void* const* d_in, const int* in_sizes, int n_in,
                              void* d_out, int out_size, void* d_ws, size_t ws_size,
                              hipStream_t stream) {
  const float* x      = (const float*)d_in[0];
  const float* w_u_w  = (const float*)d_in[1];
  const float* w_u_b  = (const float*)d_in[2];
  const float* w_v_w  = (const float*)d_in[3];
  const float* w_v_b  = (const float*)d_in[4];
  const float* proj_w = (const float*)d_in[5];
  const float* proj_b = (const float*)d_in[6];
  float* out = (float*)d_out;

  const size_t xN = (size_t)B_ * T_ * C_;   // 3,145,728
  const size_t wN = (size_t)C_ * C_;        //   589,824

  _Float16* x_h  = (_Float16*)d_ws;
  _Float16* u_h  = x_h  + xN;
  _Float16* v_h  = u_h  + xN;
  _Float16* y_h  = v_h  + xN;
  _Float16* wu_h = y_h  + xN;
  _Float16* wv_h = wu_h + wN;
  _Float16* wp_h = wv_h + wN;
  // total: 4*xN + 3*wN halves = ~28.7 MB of workspace

  // stage 1: down-convert inputs to f16
  cvt_f32_f16_kernel<<<(int)((xN + 255) / 256), 256, 0, stream>>>(x, x_h, (int)xN);
  cvt_f32_f16_kernel<<<(int)((wN + 255) / 256), 256, 0, stream>>>(w_u_w, wu_h, (int)wN);
  cvt_f32_f16_kernel<<<(int)((wN + 255) / 256), 256, 0, stream>>>(w_v_w, wv_h, (int)wN);
  cvt_f32_f16_kernel<<<(int)((wN + 255) / 256), 256, 0, stream>>>(proj_w, wp_h, (int)wN);

  const int M = B_ * T_;  // 4096
  dim3 ggrid(M / 128, C_ / 64);  // (32, 12)

  // stage 2: u = x @ Wu^T + bu ; v = x @ Wv^T + bv
  wmma_gemm_kernel<<<ggrid, 128, 0, stream>>>(x_h, wu_h, w_u_b, u_h, nullptr, M, C_, C_);
  wmma_gemm_kernel<<<ggrid, 128, 0, stream>>>(x_h, wv_h, w_v_b, v_h, nullptr, M, C_, C_);

  // stage 3: causal attention, y = softmax(u x^T * scale) v
  attn_kernel<<<B_ * H_ * (T_ / 64), 128, 0, stream>>>(x_h, u_h, v_h, y_h);

  // stage 4: out = y @ Wp^T + bp (f32 output)
  wmma_gemm_kernel<<<ggrid, 128, 0, stream>>>(y_h, wp_h, proj_b, nullptr, out, M, C_, C_);
}
